// DCGRUCell_2671469658627
// MI455X (gfx1250) — matrix operand order repaired and checked by reading, not compiled, not run on hardware
//
#include <hip/hip_runtime.h>

// DCGRU cell for gfx1250 (MI455X, wave32, WMMA).
// inputs: [N,64] f32, states: [N,128] f32, edge_index: [2,E] i32,
// W_ru: [192,256], b_ru: [256], W_c: [192,128], b_c: [128]
// out: (new_state, new_state) concatenated -> 2*N*128 f32.

#define IN_F   64
#define HID_F  128
#define K_F    192   // IN_F + HID_F
#define KP     (K_F + 8)   // padded LDS K stride (400 B, 16-B aligned)

typedef __attribute__((ext_vector_type(16))) _Float16 v16h;
typedef __attribute__((ext_vector_type(8)))  _Float16 v8h;
typedef __attribute__((ext_vector_type(8)))  float    v8f;

// ---------------------------------------------------------------- zero
__global__ void zero_kernel(float* __restrict__ p, long long n) {
    long long i = (long long)blockIdx.x * blockDim.x + threadIdx.x;
    if (i < n) p[i] = 0.0f;
}

// ---------------------------------------------------------------- scatter: ru phase
// One edge per 192-thread block; lane f accumulates feature f at dst.
__global__ void scatter_ru_kernel(const float* __restrict__ inputs,
                                  const float* __restrict__ states,
                                  const int* __restrict__ src,
                                  const int* __restrict__ dst,
                                  float* __restrict__ agg,   // [N,192]
                                  float* __restrict__ deg,   // [N]
                                  int nEdges) {
    int e = blockIdx.x;
    if (e >= nEdges) return;
    int s = src[e];
    int d = dst[e];
    int f = threadIdx.x;                       // 0..191
    float v = (f < IN_F) ? inputs[(long long)s * IN_F + f]
                         : states[(long long)s * HID_F + (f - IN_F)];
    atomicAdd(&agg[(long long)d * K_F + f], v);
    if (f == 0) atomicAdd(&deg[d], 1.0f);
}

// ---------------------------------------------------------------- scatter: candidate phase
// Same, but hidden part is r*states with r = ru[:, 0:128].
__global__ void scatter_c_kernel(const float* __restrict__ inputs,
                                 const float* __restrict__ states,
                                 const float* __restrict__ ru,   // [N,256]
                                 const int* __restrict__ src,
                                 const int* __restrict__ dst,
                                 float* __restrict__ agg,
                                 float* __restrict__ deg,
                                 int nEdges) {
    int e = blockIdx.x;
    if (e >= nEdges) return;
    int s = src[e];
    int d = dst[e];
    int f = threadIdx.x;
    float v;
    if (f < IN_F) {
        v = inputs[(long long)s * IN_F + f];
    } else {
        int j = f - IN_F;
        float r = ru[(long long)s * (2 * HID_F) + j];   // r gate
        v = r * states[(long long)s * HID_F + j];
    }
    atomicAdd(&agg[(long long)d * K_F + f], v);
    if (f == 0) atomicAdd(&deg[d], 1.0f);
}

// ---------------------------------------------------------------- WMMA GEMM + activation
// out[N,OUT] = act( (agg/deg) @ W[K_F,OUT] + bias ),  ACT 0=sigmoid, 1=tanh
// Block: 128 threads = 4 waves; block computes a 32-row x 64-col tile.
// Each wave owns a 16-col strip and computes two 16x16 WMMA tiles (rows 0-15
// and 16-31) sharing one B fragment. K=192 consumed as 6 steps of K=32.
// B tile is staged TRANSPOSED in LDS so fragment reads are contiguous b128s.
template <int OUT, int ACT>
__global__ void gemm_act_kernel(const float* __restrict__ agg,
                                const float* __restrict__ deg,
                                const float* __restrict__ W,     // [K_F, OUT]
                                const float* __restrict__ bias,  // [OUT]
                                float* __restrict__ out,         // [N, OUT]
                                int nRows) {
    __shared__ _Float16 As[32][KP];       // [row][k]
    __shared__ _Float16 BsT[64][KP];      // [col][k]  (transposed)

    const int tid     = threadIdx.x;      // 0..127
    const int wave    = tid >> 5;         // 0..3
    const int lane    = tid & 31;
    const int rowBase = blockIdx.x * 32;
    const int colBase = blockIdx.y * 64;

    // Hint: pull the weight strip toward the caches (global_prefetch_b8).
    __builtin_prefetch(&W[(long long)(tid & (K_F - 1)) * OUT + colBase], 0, 1);

    // Stage A tile (32 x 192 f32 -> f16, mean-normalized by degree).
    for (int i = tid; i < 32 * K_F; i += 128) {
        int r = i / K_F, k = i - r * K_F;
        int row = rowBase + r;
        float a = 0.0f;
        if (row < nRows) {
            float dg  = deg[row];
            float inv = 1.0f / fmaxf(dg, 1.0f);
            a = agg[(long long)row * K_F + k] * inv;
        }
        As[r][k] = (_Float16)a;
    }
    // Stage B tile transposed (192 x 64 f32 -> f16), coalesced global reads.
    for (int i = tid; i < K_F * 64; i += 128) {
        int k = i >> 6, c = i & 63;
        BsT[c][k] = (_Float16)W[(long long)k * OUT + colBase + c];
    }
    __syncthreads();

    // Wave32 WMMA fragment coordinates (ISA 7.12.2):
    //  A 16x32 f16: lane M = lane&15; elems 0..7 -> K = half*8 + e,
    //               elems 8..15 -> K = 16 + half*8 + (e-8)
//  B 32x16 f16: lane N = lane&15; same K mapping as A
    //  C/D 16x16 f32: lane N = lane&15; vgpr r -> M = r + half*8
    const int half = lane >> 4;
    const int l15  = lane & 15;
    const int wcol = wave * 16;

    v8f acc0 = {};   // rows rowBase + 0..15
    v8f acc1 = {};   // rows rowBase + 16..31
#pragma unroll
    for (int kk = 0; kk < K_F / 32; ++kk) {
        const int k0  = kk * 32;
        const int akb = k0 + half * 8;
        const int bkb = k0 + half * 8;

        v8h a0l = *(const v8h*)&As[l15][akb];
        v8h a0h = *(const v8h*)&As[l15][akb + 16];
        v8h a1l = *(const v8h*)&As[l15 + 16][akb];
        v8h a1h = *(const v8h*)&As[l15 + 16][akb + 16];
        v8h bl  = *(const v8h*)&BsT[wcol + l15][bkb];
        v8h bh  = *(const v8h*)&BsT[wcol + l15][bkb + 16];

        v16h a0 = __builtin_shufflevector(a0l, a0h, 0,1,2,3,4,5,6,7,8,9,10,11,12,13,14,15);
        v16h a1 = __builtin_shufflevector(a1l, a1h, 0,1,2,3,4,5,6,7,8,9,10,11,12,13,14,15);
        v16h b  = __builtin_shufflevector(bl,  bh,  0,1,2,3,4,5,6,7,8,9,10,11,12,13,14,15);

        acc0 = __builtin_amdgcn_wmma_f32_16x16x32_f16(false, a0, false, b,
                                                      (short)0, acc0, false, false);
        acc1 = __builtin_amdgcn_wmma_f32_16x16x32_f16(false, a1, false, b,
                                                      (short)0, acc1, false, false);
    }

    // Epilogue: bias + activation + store.
    const int n  = colBase + wcol + l15;
    const float bv = bias[n];
#pragma unroll
    for (int r = 0; r < 8; ++r) {
        const int m = r + half * 8;
        {
            const int row = rowBase + m;
            if (row < nRows) {
                float v = acc0[r] + bv;
                if (ACT == 0) v = 1.0f / (1.0f + __expf(-v));
                else          v = tanhf(v);
                out[(long long)row * OUT + n] = v;
            }
        }
        {
            const int row = rowBase + 16 + m;
            if (row < nRows) {
                float v = acc1[r] + bv;
                if (ACT == 0) v = 1.0f / (1.0f + __expf(-v));
                else          v = tanhf(v);
                out[(long long)row * OUT + n] = v;
            }
        }
    }
}

// ---------------------------------------------------------------- gated combine
// new = u*h + (1-u)*c ; write both tuple copies. c aliases out+total safely
// (each thread reads its own c[i] before writing out[total+i]).
__global__ void combine_kernel(const float* __restrict__ states,
                               const float* __restrict__ ru,   // [N,256]
                               const float* __restrict__ c,    // [N,128]
                               float* __restrict__ out,        // [2*N*128]
                               long long total) {              // N*128
    long long i = (long long)blockIdx.x * blockDim.x + threadIdx.x;
    if (i >= total) return;
    long long row = i >> 7;          // /128
    int       j   = (int)(i & 127);
    float u  = ru[row * (2 * HID_F) + HID_F + j];   // u gate
    float cv = c[i];
    float nv = u * states[i] + (1.0f - u) * cv;
    out[i]         = nv;
    out[total + i] = nv;
}

// ---------------------------------------------------------------- launcher
extern "C" void kernel_launch(void* const* d_in, const int* in_sizes, int n_in,
                              void* d_out, int out_size, void* d_ws, size_t ws_size,
                              hipStream_t stream) {
    const float* inputs = (const float*)d_in[0];
    const float* states = (const float*)d_in[1];
    const int*   eidx   = (const int*)  d_in[2];
    const float* W_ru   = (const float*)d_in[3];
    const float* b_ru   = (const float*)d_in[4];
    const float* W_c    = (const float*)d_in[5];
    const float* b_c    = (const float*)d_in[6];
    float* out = (float*)d_out;

    const int N = in_sizes[0] / IN_F;
    const int E = in_sizes[2] / 2;
    const int* src = eidx;
    const int* dst = eidx + E;

    // Workspace layout (floats): agg[N*192] | deg[N] | ru[N*256]
    float* ws  = (float*)d_ws;
    float* agg = ws;
    float* deg = ws + (long long)N * K_F;
    float* ru  = ws + (long long)N * (K_F + 1);
    // Candidate c lives in the second tuple-copy region of d_out (safe alias).
    long long total = (long long)N * HID_F;
    float* c = out + total;

    const long long nz = (long long)N * (K_F + 1);
    const int ZB = 256;
    dim3 zgrid((unsigned)((nz + ZB - 1) / ZB));
    dim3 mgrid((unsigned)((N + 31) / 32), 4);   // 256 output cols / 64
    dim3 cgrid((unsigned)((N + 31) / 32), 2);   // 128 output cols / 64
    dim3 kgrid((unsigned)((total + ZB - 1) / ZB));

    // ---- Phase 1: r,u gates ----
    zero_kernel<<<zgrid, ZB, 0, stream>>>(agg, nz);
    scatter_ru_kernel<<<E, K_F, 0, stream>>>(inputs, states, src, dst, agg, deg, E);
    gemm_act_kernel<2 * HID_F, 0><<<mgrid, 128, 0, stream>>>(agg, deg, W_ru, b_ru, ru, N);

    // ---- Phase 2: candidate ----
    zero_kernel<<<zgrid, ZB, 0, stream>>>(agg, nz);
    scatter_c_kernel<<<E, K_F, 0, stream>>>(inputs, states, ru, src, dst, agg, deg, E);
    gemm_act_kernel<HID_F, 1><<<cgrid, 128, 0, stream>>>(agg, deg, W_c, b_c, c, N);

    // ---- Combine + write both tuple copies ----
    combine_kernel<<<kgrid, ZB, 0, stream>>>(states, ru, c, out, total);
}
